// CausalFlowModel_20607253086636
// MI455X (gfx1250) — matrix operand order repaired and checked by reading, not compiled, not run on hardware
//
#include <hip/hip_runtime.h>
#include <hip/hip_bf16.h>
#include <math.h>

typedef __attribute__((ext_vector_type(16))) __bf16 v16bf;
typedef __attribute__((ext_vector_type(8)))  float  v8f;

#define B_    256
#define T_    512
#define S_    8
#define CD1   5        // CD + 1
#define RNN_  512
#define H_    520      // RNN + S
#define E_    1024
#define D_    1040
#define G4    2080     // 4 * H
#define KT_   18       // K tiles of 32: 17 cover h (520 -> 544), tile 17 carries u_t (5 -> 32)
#define JT_   33       // h-column tiles of 16: cover j = 0..527 (520 padded)
#define NTP_  132      // JT_ * 4 gates
#define HSTR  521      // LDS row stride (bank-conflict pad)
#define NWAVE 11       // 11 waves, 3 j-tiles each (stride NWAVE), perfectly balanced
#define NTHR  (NWAVE * 32)

// ---------------------------------------------------------------------------
// Pack [W_hh ; zeros ; W_ih] into bf16 WMMA B-fragment tiles.
// Tile (kt, ntp): 32 lanes x 16 halves, per ISA 7.12.2 (32x16 16-bit B):
//   element (k_local, n_local) -> lane = 16*((k_local/8)%2) + n_local, half q = k_local%8 + 8*(k_local/16).
// Column remap: ntp = 4*jt + gate, n_local -> original col = gate*H_ + (16*jt + n_local).
// ---------------------------------------------------------------------------
__global__ void pack_weights(const float* __restrict__ W_ih,
                             const float* __restrict__ W_hh,
                             __bf16* __restrict__ packedB) {
    int tile = blockIdx.x;               // 0 .. KT_*NTP_-1
    int kt   = tile / NTP_;
    int ntp  = tile % NTP_;
    int jt   = ntp >> 2;
    int gate = ntp & 3;
    for (int e = threadIdx.x; e < 512; e += 256) {
        int l = e >> 4, q = e & 15;
        int k = kt * 32 + ((q < 8) ? (8 * (l >> 4) + q) : (16 + 8 * (l >> 4) + (q - 8)));     // augmented K index
        int j = jt * 16 + (l & 15);              // h column
        float v = 0.f;
        if (j < H_) {
            int col = gate * H_ + j;
            if (k < H_)                     v = W_hh[(size_t)k * G4 + col];
            else if (k >= 544 && k < 549)   v = W_ih[(size_t)(k - 544) * G4 + col];
        }
        packedB[(size_t)tile * 512 + e] = (__bf16)v;
    }
}

// ---------------------------------------------------------------------------
// Persistent LSTM scan. One block = 16 batch rows (WMMA M=16), 11 waves.
//  - h/c state resident in LDS for all 512 steps; h_seq never materialized
//    (roll/interp/gather fused via the t == len-1 capture).
//  - A-fragments double-buffered in LDS; the state-update phase writes next
//    step's fragments directly in ISA A-layout -> no rebuild pass.
//  - rnn_input slice preloaded to LDS as bf16 -> no global latency per step.
//  - 2 barriers per step.
//  - unroll pragmas keep exactly one j-tile (4 accumulators) in flight so the
//    B-fragment stream stays global_load_b128 -> v_wmma with no scratch spill.
// ---------------------------------------------------------------------------
__global__ __launch_bounds__(NTHR, 1)
void lstm_scan(const float* __restrict__ rnn_input,   // [B][T][CD1]
               const float* __restrict__ deltas,      // [B][T]
               const int*   __restrict__ h_lens,      // [B]
               const __bf16* __restrict__ packedB,    // [KT_][NTP_][512]
               const float* __restrict__ b_lstm,      // [G4]
               const float* __restrict__ h0,          // [B][H_]
               float* __restrict__ last)              // [B][H_]
{
    __shared__ float h_s[16][HSTR];
    __shared__ float c_s[16][HSTR];
    __shared__ __attribute__((aligned(32))) __bf16 afrag[2][KT_ * 512];
    __shared__ float bias_s[G4];
    __shared__ __bf16 u_all[16 * T_ * CD1];   // block's rnn_input slice (contiguous)
    __shared__ int   len_s[16];

    const int tid  = threadIdx.x;
    const int lane = tid & 31;
    const int wave = tid >> 5;
    const int m0   = blockIdx.x * 16;

    // ---- one-time init ----
    for (int i = tid; i < 16 * H_; i += NTHR) {
        int m = i / H_, j = i % H_;
        h_s[m][j] = h0[(size_t)(m0 + m) * H_ + j];
        c_s[m][j] = 0.f;
    }
    for (int i = tid; i < G4; i += NTHR) bias_s[i] = b_lstm[i];
    if (tid < 16) len_s[tid] = h_lens[m0 + tid];
    // rnn_input rows m0..m0+15 are one contiguous run of 16*T_*CD1 floats
    {
        const float* src = rnn_input + (size_t)m0 * T_ * CD1;
        for (int i = tid; i < 16 * T_ * CD1; i += NTHR)
            u_all[i] = (__bf16)src[i];
    }
    // buffer 0: A-fragments of h0 (ISA 7.12.2 16x32 A layout); buffer 1: zeros
    for (int idx = tid; idx < KT_ * 512; idx += NTHR) {
        int kt = idx >> 9, e = idx & 511;
        int l = e >> 4, q = e & 15;
        int m = l & 15, g = l >> 4;
        int v = q >> 1, hw = q & 1;
        int kl = (v < 4) ? (8 * g + 2 * v + hw)
                         : (16 + 8 * g + 2 * (v - 4) + hw);
        int k = kt * 32 + kl;
        float val = (k < H_) ? h0[(size_t)(m0 + (size_t)m) * H_ + k] : 0.f;
        afrag[0][idx] = (__bf16)val;
        afrag[1][idx] = (__bf16)0.f;
    }

#pragma unroll 1
    for (int t = 0; t < T_; ++t) {
        const int rb = t & 1, wb = rb ^ 1;
        __syncthreads();                        // prev step's frag/state writes done
        if (tid < 16 * CD1) {
            // u_t into read buffer, tile 17 (kl = 0..4 -> lane m, half kl)
            int m = tid / CD1, k = tid % CD1;
            afrag[rb][17 * 512 + m * 16 + k] = u_all[(m * T_ + t) * CD1 + k];
        }
        __syncthreads();                        // read buffer complete

#pragma unroll 1
        for (int jt = wave; jt < JT_; jt += NWAVE) {
            const int n  = lane & 15;
            const int j  = jt * 16 + n;
            float bi = (j < H_) ? bias_s[j]          : 0.f;
            float bff= (j < H_) ? bias_s[H_ + j]     : 0.f;
            float bg = (j < H_) ? bias_s[2 * H_ + j] : 0.f;
            float bo = (j < H_) ? bias_s[3 * H_ + j] : 0.f;
            v8f ai = {bi,bi,bi,bi,bi,bi,bi,bi};
            v8f af = {bff,bff,bff,bff,bff,bff,bff,bff};
            v8f ag = {bg,bg,bg,bg,bg,bg,bg,bg};
            v8f ao = {bo,bo,bo,bo,bo,bo,bo,bo};

            for (int kt = 0; kt < KT_; ++kt) {
                v16bf a = *(const v16bf*)(&afrag[rb][kt * 512 + lane * 16]);
                const __bf16* bp =
                    packedB + ((size_t)kt * NTP_ + (size_t)jt * 4) * 512 + lane * 16;
                v16bf b0 = *(const v16bf*)(bp);
                v16bf b1 = *(const v16bf*)(bp + 512);
                v16bf b2 = *(const v16bf*)(bp + 1024);
                v16bf b3 = *(const v16bf*)(bp + 1536);
                ai = __builtin_amdgcn_wmma_f32_16x16x32_bf16(false, a, false, b0, (short)0, ai, false, false);
                af = __builtin_amdgcn_wmma_f32_16x16x32_bf16(false, a, false, b1, (short)0, af, false, false);
                ag = __builtin_amdgcn_wmma_f32_16x16x32_bf16(false, a, false, b2, (short)0, ag, false, false);
                ao = __builtin_amdgcn_wmma_f32_16x16x32_bf16(false, a, false, b3, (short)0, ao, false, false);
            }

            if (j < H_) {
                const int mg  = (lane >> 4) * 8;   // C/D layout: lane group -> rows 0-7 / 8-15
                const int kt2 = j >> 5;
                const int kl  = j & 31;
                const int g2  = (kl >> 3) & 1;
                const int q2  = (kl & 7) | ((kl >> 1) & 8);
                for (int v = 0; v < 8; ++v) {
                    int m = mg + v;
                    float si = 1.f / (1.f + expf(-ai[v]));
                    float sf = 1.f / (1.f + expf(-af[v]));
                    float tg = tanhf(ag[v]);
                    float so = 1.f / (1.f + expf(-ao[v]));
                    float cn = sf * c_s[m][j] + si * tg;
                    float hn = so * tanhf(cn);
                    float h_old = h_s[m][j];       // == y_{t-1} (or h0_stack at t==0)
                    int len = len_s[m];
                    if (t == len - 1) {
                        float d = deltas[(size_t)(m0 + m) * T_ + t];
                        last[(size_t)(m0 + m) * H_ + j] =
                            (1.f - d) * h_old + d * hn;
                    }
                    bool upd = (t < len);
                    float hkeep = upd ? hn : h_old;
                    // write next step's A fragment slot (held rows re-written)
                    afrag[wb][kt2 * 512 + (16 * g2 + m) * 16 + q2] = (__bf16)hkeep;
                    if (upd) { c_s[m][j] = cn; h_s[m][j] = hn; }
                }
            }
        }
    }
}

// ---------------------------------------------------------------------------
// Small dense layers (<1% of FLOPs): thread-per-output f32 GEMM + bias + tanh.
// ---------------------------------------------------------------------------
__global__ void gemm_bias_act(const float* __restrict__ A, int lda,
                              const float* __restrict__ Bm, int ldb,
                              const float* __restrict__ bias,
                              float* __restrict__ Dst, int ldd,
                              int M, int N, int K, int do_tanh) {
    int n = blockIdx.x * blockDim.x + threadIdx.x;
    int m = blockIdx.y * blockDim.y + threadIdx.y;
    if (m >= M || n >= N) return;
    float s = bias ? bias[n] : 0.f;
    for (int k = 0; k < K; ++k)
        s += A[(size_t)m * lda + k] * Bm[(size_t)k * ldb + n];
    Dst[(size_t)m * ldd + n] = do_tanh ? tanhf(s) : s;
}

__global__ void copy_x_into_h0(const float* __restrict__ x, float* __restrict__ h0) {
    int idx = blockIdx.x * blockDim.x + threadIdx.x;
    if (idx >= B_ * S_) return;
    int m = idx / S_, c = idx % S_;
    h0[(size_t)m * H_ + c] = x[(size_t)m * S_ + c];
}

extern "C" void kernel_launch(void* const* d_in, const int* in_sizes, int n_in,
                              void* d_out, int out_size, void* d_ws, size_t ws_size,
                              hipStream_t stream) {
    (void)in_sizes; (void)n_in; (void)out_size; (void)ws_size;
    const float* x      = (const float*)d_in[0];
    const float* rnn_in = (const float*)d_in[1];
    const float* deltas = (const float*)d_in[2];
    const int*   h_lens = (const int*)d_in[3];
    const float* enc_w1 = (const float*)d_in[4];
    const float* enc_b1 = (const float*)d_in[5];
    const float* enc_w2 = (const float*)d_in[6];
    const float* enc_b2 = (const float*)d_in[7];
    const float* enc_w3 = (const float*)d_in[8];
    const float* enc_b3 = (const float*)d_in[9];
    const float* W_ih   = (const float*)d_in[10];
    const float* W_hh   = (const float*)d_in[11];
    const float* b_lstm = (const float*)d_in[12];
    const float* dec_w1 = (const float*)d_in[13];
    const float* dec_b1 = (const float*)d_in[14];
    const float* dec_w2 = (const float*)d_in[15];
    const float* dec_b2 = (const float*)d_in[16];
    const float* dec_w3 = (const float*)d_in[17];
    const float* dec_b3 = (const float*)d_in[18];
    float* out = (float*)d_out;

    char* wsp = (char*)d_ws;
    auto alloc = [&](size_t bytes) {
        char* p = wsp;
        wsp += (bytes + 255) & ~(size_t)255;
        return p;
    };
    __bf16* packedB = (__bf16*)alloc((size_t)KT_ * NTP_ * 512 * sizeof(__bf16));
    float* h0    = (float*)alloc((size_t)B_ * H_ * sizeof(float));
    float* t1    = (float*)alloc((size_t)B_ * E_ * sizeof(float));
    float* t2    = (float*)alloc((size_t)B_ * E_ * sizeof(float));
    float* lastb = (float*)alloc((size_t)B_ * H_ * sizeof(float));
    float* z1    = (float*)alloc((size_t)B_ * D_ * sizeof(float));
    float* z2    = (float*)alloc((size_t)B_ * D_ * sizeof(float));

    dim3 blk(16, 16);
    // encoder
    gemm_bias_act<<<dim3((E_ + 15) / 16, (B_ + 15) / 16), blk, 0, stream>>>(
        x, S_, enc_w1, E_, enc_b1, t1, E_, B_, E_, S_, 1);
    gemm_bias_act<<<dim3((E_ + 15) / 16, (B_ + 15) / 16), blk, 0, stream>>>(
        t1, E_, enc_w2, E_, enc_b2, t2, E_, B_, E_, E_, 1);
    gemm_bias_act<<<dim3((RNN_ + 15) / 16, (B_ + 15) / 16), blk, 0, stream>>>(
        t2, E_, enc_w3, RNN_, enc_b3, h0 + S_, H_, B_, RNN_, E_, 0);
    copy_x_into_h0<<<(B_ * S_ + 255) / 256, 256, 0, stream>>>(x, h0);
    // weight packing for WMMA B fragments
    pack_weights<<<KT_ * NTP_, 256, 0, stream>>>(W_ih, W_hh, packedB);
    // fused LSTM scan + roll/interp/gather
    lstm_scan<<<B_ / 16, NTHR, 0, stream>>>(rnn_in, deltas, h_lens, packedB,
                                            b_lstm, h0, lastb);
    // decoder
    gemm_bias_act<<<dim3((D_ + 15) / 16, (B_ + 15) / 16), blk, 0, stream>>>(
        lastb, H_, dec_w1, D_, dec_b1, z1, D_, B_, D_, H_, 1);
    gemm_bias_act<<<dim3((D_ + 15) / 16, (B_ + 15) / 16), blk, 0, stream>>>(
        z1, D_, dec_w2, D_, dec_b2, z2, D_, B_, D_, D_, 1);
    gemm_bias_act<<<dim3((S_ + 15) / 16, (B_ + 15) / 16), blk, 0, stream>>>(
        z2, D_, dec_w3, S_, dec_b3, out, S_, B_, S_, D_, 0);
}
